// ResidualRecursiveNN_46351287058653
// MI455X (gfx1250) — compile-verified, matrix operand order checked by reading, not run on hardware
//
#include <hip/hip_runtime.h>
#include <hip/hip_bf16.h>
#include <stdint.h>

#define TDEPTH 15
#define EMBED  256
#define NNODES 65535   // 2^16 - 1
#define BATCH  4

typedef __attribute__((ext_vector_type(16))) __bf16 v16bf;
typedef __attribute__((ext_vector_type(8)))  float  v8f;

__device__ __forceinline__ uint32_t f32_to_bf16_rne(float f) {
  uint32_t u = __float_as_uint(f);
  u += 0x7fffu + ((u >> 16) & 1u);
  return u >> 16;
}

// ---------------------------------------------------------------- W -> bf16
__global__ void wconv_kernel(const float* __restrict__ W,
                             uint16_t* __restrict__ Wbf, int n) {
  int i = blockIdx.x * blockDim.x + threadIdx.x;
  if (i < n) Wbf[i] = (uint16_t)f32_to_bf16_rne(W[i]);
}

// ------------------------------------------------------- leaf gather + relu
__global__ void leaf_kernel(const int* __restrict__ wid,
                            const float* __restrict__ emb,
                            float* __restrict__ h) {
  int t   = blockIdx.x * 256 + threadIdx.x;   // one float4 per thread
  int row = t >> 6;                           // 64 float4 per 256-f32 row
  int q   = t & 63;
  int b   = row >> TDEPTH;
  int i   = row & ((1 << TDEPTH) - 1);
  int w   = wid[b * NNODES + ((1 << TDEPTH) - 1) + i];
  float4 v = *(const float4*)(emb + (size_t)w * EMBED + q * 4);
  float4 o;
  o.x = fmaxf(v.x, 0.f); o.y = fmaxf(v.y, 0.f);
  o.z = fmaxf(v.z, 0.f); o.w = fmaxf(v.w, 0.f);
  *(float4*)(h + (size_t)row * EMBED + q * 4) = o;
}

// ------------------------------------------------------------- level GEMM
// Out[r, e] = relu( sum_f X[r,f] * W[e,f] + bias[e] (+ g[2r,e] + g[2r+1,e]) )
// X[r] = [ h_in[2r*256 ..], h_in[(2r+1)*256 ..], emb[wid(r)] ]   (768 wide)
// 32-row M-tile per workgroup; each wave owns 16 output cols and two
// accumulators (rows 0-15 / 16-31) so every B fragment feeds two WMMAs.
#define LDS_STRIDE 1552   // 768*2 + 16 pad: conflict-free b128 A loads

__global__ __launch_bounds__(512)
void level_kernel(const float* __restrict__ h_in, float* __restrict__ h_out,
                  const int* __restrict__ word_ids, const float* __restrict__ emb,
                  const uint16_t* __restrict__ Wbf, const float* __restrict__ Wb,
                  float* __restrict__ g, int d, int flags /*1=write_g 2=add_g*/,
                  int Mrows) {
  __shared__ __align__(16) uint8_t smem[32 * LDS_STRIDE];
  const int t     = threadIdx.x;
  const int mtile = blockIdx.x;
  const int n     = 1 << d;
  const int s     = n - 1;

  // ---- stage X tile (32 rows x 768) into LDS as bf16 ----
  {
    const int sub = t & 31;
#pragma unroll
    for (int half = 0; half < 2; ++half) {
      const int rl = (t >> 5) + 16 * half;            // local row 0..31
      const int r  = mtile * 32 + rl;
      uint32_t* dst = (uint32_t*)(smem + rl * LDS_STRIDE);
      if (r < Mrows) {
        const int b = r >> d;
        const int i = r & (n - 1);
        const int w = word_ids[b * NNODES + s + i];
        const float* hl = h_in + (size_t)r * 512;     // l then r, contiguous
        const float* er = emb + (size_t)w * EMBED;
#pragma unroll
        for (int j = 0; j < 12; ++j) {
          int f = 2 * sub + 64 * j;                   // 0..766, pairwise
          float2 v = (f < 512) ? *(const float2*)(hl + f)
                               : *(const float2*)(er + (f - 512));
          dst[f >> 1] = f32_to_bf16_rne(v.x) | (f32_to_bf16_rne(v.y) << 16);
        }
        if (flags & 1) {                              // g[r] = l + r (fp32)
#pragma unroll
          for (int j = 0; j < 4; ++j) {
            int e = 2 * sub + 64 * j;
            float2 a = *(const float2*)(hl + e);
            float2 c = *(const float2*)(hl + 256 + e);
            float2 o; o.x = a.x + c.x; o.y = a.y + c.y;
            *(float2*)(g + (size_t)r * EMBED + e) = o;
          }
        }
      } else {
#pragma unroll
        for (int j = 0; j < 12; ++j) { int f = 2 * sub + 64 * j; dst[f >> 1] = 0; }
      }
    }
  }
  __syncthreads();

  // ---- two 16x16 output tiles per wave, K=768 via 2x24 wmma bf16 k32 ----
  const int wave    = t >> 5;
  const int lane    = t & 31;
  const int mrow    = lane & 15;
  const int hi      = lane >> 4;
  const int colbase = wave * 16;
  union Frag { uint4 q[2]; v16bf v; };
  v8f acc0 = {};
  v8f acc1 = {};
  const uint8_t* aptr0 = smem + mrow * LDS_STRIDE + hi * 16;
  const uint8_t* aptr1 = smem + (16 + mrow) * LDS_STRIDE + hi * 16;
  const uint8_t* wptr  = (const uint8_t*)Wbf +
                         (size_t)(colbase + mrow) * (768 * 2) + hi * 16;
#pragma unroll 4
  for (int k = 0; k < 24; ++k) {
    Frag A0, A1, Bf;
    Bf.q[0] = *(const uint4*)(wptr + k * 64);
    Bf.q[1] = *(const uint4*)(wptr + k * 64 + 32);
    A0.q[0] = *(const uint4*)(aptr0 + k * 64);
    A0.q[1] = *(const uint4*)(aptr0 + k * 64 + 32);
    A1.q[0] = *(const uint4*)(aptr1 + k * 64);
    A1.q[1] = *(const uint4*)(aptr1 + k * 64 + 32);
    acc0 = __builtin_amdgcn_wmma_f32_16x16x32_bf16(
        false, A0.v, false, Bf.v, (short)0, acc0, false, false);
    acc1 = __builtin_amdgcn_wmma_f32_16x16x32_bf16(
        false, A1.v, false, Bf.v, (short)0, acc1, false, false);
  }

  // ---- epilogue: bias (+ grandchild residual), relu, store ----
  const int e = colbase + mrow;
  const float bias = Wb[e];
#pragma unroll
  for (int v = 0; v < 8; ++v) {
#pragma unroll
    for (int half = 0; half < 2; ++half) {
      const int r = mtile * 32 + 16 * half + v + 8 * hi;   // C/D row mapping
      if (r < Mrows) {
        float val = (half ? acc1[v] : acc0[v]) + bias;
        if (flags & 2)
          val += g[(size_t)(2 * r) * EMBED + e] +
                 g[(size_t)(2 * r + 1) * EMBED + e];
        h_out[(size_t)r * EMBED + e] = fmaxf(val, 0.0f);
      }
    }
  }
}

// ------------------------------------------------------------------- head
__global__ void head_kernel(const float* __restrict__ root,
                            const int* __restrict__ labels,
                            const float* __restrict__ Pw,
                            const float* __restrict__ Pb,
                            float* __restrict__ out) {
  int b = threadIdx.x;
  if (b >= BATCH) return;
  float l0 = Pb[0], l1 = Pb[1];
  for (int e = 0; e < EMBED; ++e) {
    float x = root[b * EMBED + e];
    l0 = fmaf(x, Pw[e], l0);
    l1 = fmaf(x, Pw[EMBED + e], l1);
  }
  int pred = (l1 > l0) ? 1 : 0;                 // jnp.argmax: first max wins
  float m  = fmaxf(l0, l1);
  float lse = m + logf(expf(l0 - m) + expf(l1 - m));
  float loss = lse - ((labels[b] == 0) ? l0 : l1);
  out[b]         = (float)pred;
  out[BATCH + b] = loss;
}

// ---------------------------------------------------------------- launcher
extern "C" void kernel_launch(void* const* d_in, const int* in_sizes, int n_in,
                              void* d_out, int out_size, void* d_ws, size_t ws_size,
                              hipStream_t stream) {
  const int*   word_ids = (const int*)  d_in[0];
  const int*   labels   = (const int*)  d_in[1];
  const float* emb      = (const float*)d_in[2];
  const float* Ww       = (const float*)d_in[3];   // (256, 768) row-major
  const float* Wb       = (const float*)d_in[4];
  const float* Pw       = (const float*)d_in[5];
  const float* Pb       = (const float*)d_in[6];

  uint8_t* ws = (uint8_t*)d_ws;
  const size_t hA_bytes = (size_t)BATCH * 32768 * EMBED * 4;  // 128 MB
  const size_t hB_bytes = (size_t)BATCH * 16384 * EMBED * 4;  //  64 MB
  const size_t g_bytes  = (size_t)BATCH * 2048  * EMBED * 4;  //   8 MB
  float*    hA  = (float*)ws;
  float*    hB  = (float*)(ws + hA_bytes);
  float*    g   = (float*)(ws + hA_bytes + hB_bytes);
  uint16_t* Wbf = (uint16_t*)(ws + hA_bytes + hB_bytes + g_bytes);

  wconv_kernel<<<(EMBED * 768 + 255) / 256, 256, 0, stream>>>(Ww, Wbf, EMBED * 768);
  leaf_kernel<<<BATCH * 32768 * (EMBED / 4) / 256, 256, 0, stream>>>(word_ids, emb, hA);

  float* cur = hA;
  float* nxt = hB;
  for (int d = TDEPTH - 1; d >= 0; --d) {
    int Mrows  = BATCH << d;
    int mtiles = (Mrows + 31) / 32;
    int flags  = 0;
    if ((d % 10) == 0) flags |= 2;        // d = 10, 0: consume residual
    if (((d - 1) % 10) == 0) flags |= 1;  // d = 11, 1: produce residual (l+r)
    level_kernel<<<mtiles, 512, 0, stream>>>(cur, nxt, word_ids, emb,
                                             Wbf, Wb, g, d, flags, Mrows);
    float* tmp = cur; cur = nxt; nxt = tmp;
  }
  // after 15 levels, `cur` holds the B x 256 roots
  head_kernel<<<1, 32, 0, stream>>>(cur, labels, Pw, Pb, (float*)d_out);
}
